// Geo_gcn_20856361189459
// MI455X (gfx1250) — matrix operand order repaired and checked by reading, not compiled, not run on hardware
//
#include <hip/hip_runtime.h>
#include <hip/hip_bf16.h>
#include <cmath>

// Problem constants (from reference)
#define BB 16
#define CC 4
#define JJ 19
#define TT 256
#define SS 10
#define D1 64
#define D2 64
#define DSIM 128
#define OO 64
#define NCH (CC * JJ * TT)   // 19456

#define NW 6                 // waves per block (2 s-iterations cover SS=10)
#define HSTR 72              // bf16 row stride for 32x64 row-major buffers (mult of 8)
#define HCST 40              // bf16 row stride for 64x32 h-transposed buffer (mult of 8)
#define ASTR 36              // f32 row stride for att buffer (mult of 4)

typedef __attribute__((ext_vector_type(16))) __bf16 v16bf;
typedef __attribute__((ext_vector_type(8)))  __bf16 v8bf;
typedef __attribute__((ext_vector_type(8)))  float  v8f;
typedef __attribute__((ext_vector_type(4)))  float  v4f;

// ---------------------------------------------------------------------------
// WMMA operand loaders per CDNA5 ISA 7.12.2 (wave32, 16x16x32 bf16), all as
// 128-bit LDS accesses.
// A (16x32, MxK): lane m=L&15 (L>=16 -> koff=8): elements e<8 -> K=koff+e,
//                 e>=8 -> K=koff+8+e.  Two contiguous 8xbf16 runs.
// B (32x16, KxN): lane n=L&15, same K mapping as A (kb=(L&16)?8:0); two
//                 contiguous runs when B is stored N-major ([n][k]).
// C/D (16x16 f32): lane n=L&15, VGPR r -> row m = r + 8*(L>=16).
// ---------------------------------------------------------------------------

__device__ __forceinline__ v16bf ldA(const __bf16* base, int stride, int lane, int k0) {
    const int m    = lane & 15;
    const int koff = (lane & 16) ? 8 : 0;
    const __bf16* p = base + m * stride + k0 + koff;
    v8bf lo = *(const v8bf*)p;          // K = koff+0..7
    v8bf hi = *(const v8bf*)(p + 16);   // K = koff+16..23
    return __builtin_shufflevector(lo, hi, 0,1,2,3,4,5,6,7,8,9,10,11,12,13,14,15);
}

// A operand from f32 source (softmax probs), K=32 starting at 0
__device__ __forceinline__ v16bf ldA_f32(const float* base, int stride, int lane) {
    const int m    = lane & 15;
    const int koff = (lane & 16) ? 8 : 0;
    const float* p = base + m * stride + koff;
    v4f f0 = *(const v4f*)p;
    v4f f1 = *(const v4f*)(p + 4);
    v4f f2 = *(const v4f*)(p + 16);
    v4f f3 = *(const v4f*)(p + 20);
    v16bf a;
#pragma unroll
    for (int e = 0; e < 4; ++e) {
        a[e]      = (__bf16)f0[e];
        a[e + 4]  = (__bf16)f1[e];
        a[e + 8]  = (__bf16)f2[e];
        a[e + 12] = (__bf16)f3[e];
    }
    return a;
}

// B operand from N-major storage: b[e] = base[n*stride + k0 + kb + e] (e<8), base[n*stride + k0 + kb + 8 + e] (e>=8), kb = 0/8
__device__ __forceinline__ v16bf ldB(const __bf16* base, int stride, int lane, int k0) {
    const int n  = lane & 15;
    const int kb = (lane & 16) ? 8 : 0;
    const __bf16* p = base + n * stride + k0 + kb;
    v8bf lo = *(const v8bf*)p;
    v8bf hi = *(const v8bf*)(p + 16);
    return __builtin_shufflevector(lo, hi, 0,1,2,3,4,5,6,7,8,9,10,11,12,13,14,15);
}

__device__ __forceinline__ v8f wmma_bf16(v16bf a, v16bf b, v8f c) {
    return __builtin_amdgcn_wmma_f32_16x16x32_bf16(false, a, false, b, (short)0, c, false, false);
}

// 32x64x64 GEMM: A row-major (astride), B N-major (bstride), acc[mt][nt]
__device__ __forceinline__ void gemm_2x4x2(const __bf16* A, int astride,
                                           const __bf16* B, int bstride,
                                           int lane, v8f acc[2][4]) {
    v16bf vb[2][4];
#pragma unroll
    for (int kk = 0; kk < 2; ++kk)
#pragma unroll
        for (int nt = 0; nt < 4; ++nt)
            vb[kk][nt] = ldB(B + nt * 16 * bstride, bstride, lane, kk * 32);
#pragma unroll
    for (int mt = 0; mt < 2; ++mt) {
        const v16bf a0 = ldA(A + mt * 16 * astride, astride, lane, 0);
        const v16bf a1 = ldA(A + mt * 16 * astride, astride, lane, 32);
#pragma unroll
        for (int nt = 0; nt < 4; ++nt) {
            acc[mt][nt] = wmma_bf16(a0, vb[0][nt], acc[mt][nt]);
            acc[mt][nt] = wmma_bf16(a1, vb[1][nt], acc[mt][nt]);
        }
    }
}

// ---------------------------------------------------------------------------
// Kernel 0: weight prep.
//   Mc[e][c]  = sum_d ws1[d][c]*ws2[d][e]   (Mᵀ, N-major for B loads)
//   vvec[e]   = sum_d bs1[d]*ws2[d][e]
//   w2c       = bf16(w2)                    (w2 is (o,i) row-major == N-major)
//   Wc[o][c]  = W[c][o]                     (Wᵀ, N-major)
// ---------------------------------------------------------------------------
__global__ void geo_prep_kernel(const float* __restrict__ w2,
                                const float* __restrict__ ws1, const float* __restrict__ bs1,
                                const float* __restrict__ ws2, const float* __restrict__ bs2,
                                const float* __restrict__ W,
                                float* __restrict__ vvec,
                                __bf16* __restrict__ Mc, __bf16* __restrict__ w2c,
                                __bf16* __restrict__ Wc) {
    (void)bs2;
    const int tid = threadIdx.x;
    for (int i = tid; i < D2 * D2; i += 256) {
        const int e = i >> 6, c = i & 63;   // i = e*64 + c
        float m = 0.f;
        for (int d = 0; d < DSIM; ++d) m += ws1[d * D2 + c] * ws2[d * D2 + e];
        Mc[i]  = (__bf16)m;                 // Mc[e][c]
        w2c[i] = (__bf16)w2[i];             // identity copy
        Wc[i]  = (__bf16)W[c * OO + e];     // Wc[e=o][c]
    }
    if (tid < D2) {
        float sv = 0.f;
        for (int d = 0; d < DSIM; ++d) sv += bs1[d] * ws2[d * D2 + tid];
        vvec[tid] = sv;
    }
}

// ---------------------------------------------------------------------------
// Kernel 1: BatchNorm stats per channel (c,j,t): reduce over B*S = 160 elems.
// ---------------------------------------------------------------------------
__global__ void geo_bn_kernel(const float* __restrict__ x,
                              const float* __restrict__ gamma, const float* __restrict__ beta,
                              float* __restrict__ scale, float* __restrict__ shift) {
    const int ch = blockIdx.x * 256 + threadIdx.x;
    if (ch >= NCH) return;
    const int c = ch / (JJ * TT);
    const int rem = ch % (JJ * TT);
    const int j = rem / TT;
    const int t = rem % TT;
    float sum = 0.f, sq = 0.f;
    for (int b = 0; b < BB; ++b) {
        const size_t base = ((((size_t)b * CC + c) * JJ + j) * TT + t) * SS;
        for (int s = 0; s < SS; ++s) {
            const float v = x[base + s];
            sum += v; sq += v * v;
        }
    }
    const float inv_n = 1.f / (BB * SS);
    const float mean = sum * inv_n;
    const float var  = sq * inv_n - mean * mean;
    const float rs   = rsqrtf(var + 1e-5f);
    const float scl  = gamma[ch] * rs;
    scale[ch] = scl;
    shift[ch] = beta[ch] - mean * scl;
}

// ---------------------------------------------------------------------------
// Kernel 2: fused GCN layer. One block per (b,t); NW waves, 2 s-iterations.
// ---------------------------------------------------------------------------
__global__ __launch_bounds__(NW * 32) void geo_fused_kernel(
        const float* __restrict__ x,
        const float* __restrict__ w1, const float* __restrict__ b1f,
        const float* __restrict__ b2f,
        const float* __restrict__ scale, const float* __restrict__ shift,
        const float* __restrict__ vvec,
        const __bf16* __restrict__ Mc, const __bf16* __restrict__ w2c,
        const __bf16* __restrict__ Wc,
        float* __restrict__ out) {

    __shared__ alignas(16) __bf16 s_w2[D2 * D2];   // [o][i]  N-major
    __shared__ alignas(16) __bf16 s_Mc[D2 * D2];   // [e][c]  N-major
    __shared__ alignas(16) __bf16 s_Wc[D2 * D2];   // [o][c]  N-major
    __shared__ float  s_b1[D1], s_b2[D2], s_v[D2];
    __shared__ float  s_sc[CC * JJ], s_sh[CC * JJ];
    __shared__ alignas(16) __bf16 s_h  [NW][32 * HSTR];  // h row-major (j, c)
    __shared__ alignas(16) __bf16 s_hc [NW][64 * HCST];  // hᵀ (c, j)
    __shared__ alignas(16) __bf16 s_g  [NW][32 * HSTR];  // staging: h1 -> G -> agg
    __shared__ alignas(16) float  s_att[NW][32 * ASTR];  // logits -> probs

    const int tid  = threadIdx.x;
    const int t    = blockIdx.x;
    const int b    = blockIdx.y;
    const int wave = tid >> 5;
    const int lane = tid & 31;

    // --- stage shared weights / per-(c,j) BN affine for this t ---
    for (int i = tid; i < D2 * D2; i += NW * 32) {
        s_w2[i] = w2c[i];
        s_Mc[i] = Mc[i];
        s_Wc[i] = Wc[i];
    }
    if (tid < D2) { s_b1[tid] = b1f[tid]; s_b2[tid] = b2f[tid]; s_v[tid] = vvec[tid]; }
    for (int i = tid; i < CC * JJ; i += NW * 32) {
        const int c = i / JJ, j = i % JJ;
        s_sc[i] = scale[(c * JJ + j) * TT + t];
        s_sh[i] = shift[(c * JJ + j) * TT + t];
    }
    __syncthreads();

    __bf16* hb = s_h[wave];
    __bf16* hc = s_hc[wave];
    __bf16* gb = s_g[wave];
    float*  ab = s_att[wave];

    const int n  = lane & 15;
    const int mb = (lane >> 4) * 8;

#pragma unroll 1
    for (int it = 0; it < 2; ++it) {
        const int s   = it * NW + wave;
        const bool on = (s < SS);

        // ===== Phase A: h1 = relu(BN(x) @ w1ᵀ + b1)  (K=4, VALU) -> gb =====
        if (on) {
            const int o0 = lane, o1 = lane + 32;
            float w1r0[CC], w1r1[CC];
#pragma unroll
            for (int c = 0; c < CC; ++c) { w1r0[c] = w1[o0 * CC + c]; w1r1[c] = w1[o1 * CC + c]; }
            const float bb0 = s_b1[o0], bb1 = s_b1[o1];
            for (int j = 0; j < JJ; ++j) {
                float a0 = bb0, a1 = bb1;
#pragma unroll
                for (int c = 0; c < CC; ++c) {
                    const size_t xi = ((((size_t)b * CC + c) * JJ + j) * TT + t) * SS + s;
                    const float xv = x[xi] * s_sc[c * JJ + j] + s_sh[c * JJ + j];
                    a0 += xv * w1r0[c];
                    a1 += xv * w1r1[c];
                }
                gb[j * HSTR + o0] = (__bf16)fmaxf(a0, 0.f);
                gb[j * HSTR + o1] = (__bf16)fmaxf(a1, 0.f);
            }
            for (int j = JJ; j < 32; ++j) {          // zero-pad rows 19..31
                gb[j * HSTR + o0] = (__bf16)0.f;
                gb[j * HSTR + o1] = (__bf16)0.f;
            }
        }
        __syncthreads();

        // ===== Phase B: h = relu(h1 @ w2ᵀ + b2) -> hb (row-major) + hc (hᵀ) =====
        if (on) {
            v8f acc[2][4] = {};
            gemm_2x4x2(gb, HSTR, s_w2, D2, lane, acc);
#pragma unroll
            for (int mt = 0; mt < 2; ++mt)
#pragma unroll
                for (int nt = 0; nt < 4; ++nt) {
                    const float bias = s_b2[nt * 16 + n];
                    v8bf hv;
#pragma unroll
                    for (int r = 0; r < 8; ++r) {
                        const float val = fmaxf(acc[mt][nt][r] + bias, 0.f);
                        hb[(mt * 16 + mb + r) * HSTR + nt * 16 + n] = (__bf16)val;
                        hv[r] = (__bf16)val;
                    }
                    // transposed copy: row c = nt*16+n, cols j = mt*16+mb..+7 (one b128)
                    *(v8bf*)(hc + (nt * 16 + n) * HCST + mt * 16 + mb) = hv;
                }
        }
        __syncthreads();

        // ===== Phase C: G = h @ M + 1·vᵀ -> gb =====
        if (on) {
            v8f acc[2][4] = {};
            gemm_2x4x2(hb, HSTR, s_Mc, D2, lane, acc);
#pragma unroll
            for (int mt = 0; mt < 2; ++mt)
#pragma unroll
                for (int nt = 0; nt < 4; ++nt) {
                    const float vadd = s_v[nt * 16 + n];
#pragma unroll
                    for (int r = 0; r < 8; ++r)
                        gb[(mt * 16 + mb + r) * HSTR + nt * 16 + n] =
                            (__bf16)(acc[mt][nt][r] + vadd);
                }
        }
        __syncthreads();

        // ===== Phase D: logits = G @ hᵀ (32x32x64) -> ab (f32) =====
        if (on) {
            v16bf vb[2][2];
#pragma unroll
            for (int kk = 0; kk < 2; ++kk)
#pragma unroll
                for (int nt = 0; nt < 2; ++nt)
                    vb[kk][nt] = ldB(hb + nt * 16 * HSTR, HSTR, lane, kk * 32);
#pragma unroll
            for (int mt = 0; mt < 2; ++mt) {
                const v16bf a0 = ldA(gb + mt * 16 * HSTR, HSTR, lane, 0);
                const v16bf a1 = ldA(gb + mt * 16 * HSTR, HSTR, lane, 32);
#pragma unroll
                for (int nt = 0; nt < 2; ++nt) {
                    v8f acc = {};
                    acc = wmma_bf16(a0, vb[0][nt], acc);
                    acc = wmma_bf16(a1, vb[1][nt], acc);
#pragma unroll
                    for (int r = 0; r < 8; ++r)
                        ab[(mt * 16 + mb + r) * ASTR + nt * 16 + n] = acc[r];
                }
            }
        }
        __syncthreads();

        // ===== Phase E: softmax over k (valid cols 0..18; zero 19..31) =====
        if (on && lane < JJ) {
            float* row = ab + lane * ASTR;
            float mx = row[0];
            for (int k = 1; k < JJ; ++k) mx = fmaxf(mx, row[k]);
            float sum = 0.f;
            float ex[JJ];
            for (int k = 0; k < JJ; ++k) { ex[k] = __expf(row[k] - mx); sum += ex[k]; }
            const float inv = 1.f / sum;
            for (int k = 0; k < JJ; ++k) row[k] = ex[k] * inv;
            for (int k = JJ; k < 32; ++k) row[k] = 0.f;
        }
        __syncthreads();

        // ===== Phase F: agg = att @ h (32x64x32) -> gb =====
        if (on) {
            v16bf vb[4];
#pragma unroll
            for (int nt = 0; nt < 4; ++nt)
                vb[nt] = ldB(hc + nt * 16 * HCST, HCST, lane, 0);
#pragma unroll
            for (int mt = 0; mt < 2; ++mt) {
                const v16bf a0 = ldA_f32(ab + mt * 16 * ASTR, ASTR, lane);
#pragma unroll
                for (int nt = 0; nt < 4; ++nt) {
                    v8f acc = {};
                    acc = wmma_bf16(a0, vb[nt], acc);
#pragma unroll
                    for (int r = 0; r < 8; ++r)
                        gb[(mt * 16 + mb + r) * HSTR + nt * 16 + n] = (__bf16)acc[r];
                }
            }
        }
        __syncthreads();

        // ===== Phase G: out = agg @ W -> global =====
        if (on) {
            v8f acc[2][4] = {};
            gemm_2x4x2(gb, HSTR, s_Wc, D2, lane, acc);
#pragma unroll
            for (int mt = 0; mt < 2; ++mt)
#pragma unroll
                for (int nt = 0; nt < 4; ++nt) {
                    const int o = nt * 16 + n;
#pragma unroll
                    for (int r = 0; r < 8; ++r) {
                        const int j = mt * 16 + mb + r;
                        if (j < JJ)
                            out[(((size_t)b * OO + o) * JJ + j) * (TT * SS) + t * SS + s] =
                                acc[mt][nt][r];
                    }
                }
        }
        __syncthreads();
    }
}

// ---------------------------------------------------------------------------
extern "C" void kernel_launch(void* const* d_in, const int* in_sizes, int n_in,
                              void* d_out, int out_size, void* d_ws, size_t ws_size,
                              hipStream_t stream) {
    (void)in_sizes; (void)n_in; (void)out_size; (void)ws_size;
    const float* x     = (const float*)d_in[0];
    const float* gamma = (const float*)d_in[1];
    const float* beta  = (const float*)d_in[2];
    const float* w1    = (const float*)d_in[3];
    const float* b1    = (const float*)d_in[4];
    const float* w2    = (const float*)d_in[5];
    const float* b2    = (const float*)d_in[6];
    const float* ws1   = (const float*)d_in[7];
    const float* bs1   = (const float*)d_in[8];
    const float* ws2   = (const float*)d_in[9];
    const float* bs2   = (const float*)d_in[10];
    const float* W     = (const float*)d_in[11];
    float* out = (float*)d_out;

    // workspace layout
    float*  scale = (float*)d_ws;
    float*  shift = scale + NCH;
    float*  vvec  = shift + NCH;
    __bf16* Mc    = (__bf16*)(vvec + 64);
    __bf16* w2c   = Mc + D2 * D2;
    __bf16* Wc    = w2c + D2 * D2;

    geo_prep_kernel<<<1, 256, 0, stream>>>(w2, ws1, bs1, ws2, bs2, W, vvec, Mc, w2c, Wc);
    geo_bn_kernel<<<(NCH + 255) / 256, 256, 0, stream>>>(x, gamma, beta, scale, shift);
    geo_fused_kernel<<<dim3(TT, BB), NW * 32, 0, stream>>>(x, w1, b1, b2, scale, shift,
                                                           vvec, Mc, w2c, Wc, out);
}
